// ContrastiveLoss_29180007809100
// MI455X (gfx1250) — matrix operand order repaired and checked by reading, not compiled, not run on hardware
//
#include <hip/hip_runtime.h>
#include <hip/hip_bf16.h>

// ---------------------------------------------------------------------------
// NT-Xent contrastive loss, fused:
//   1) nt_normalize: L2-normalize 8192 rows -> f16 Z [8192 x 512] in workspace.
//   2) nt_sim: fused Z Z^T GEMM (v_wmma_f32_16x16x32_f16) + exp/row-sum
//      epilogue. B column tiles async-staged to LDS
//      (global_load_async_to_lds_b128 / s_wait_asynccnt, double buffered);
//      A register-resident; j-outer/k-inner WMMA chains with 2-deep B
//      rotation so ds_load latency hides behind WMMA latency.
//   3) nt_finalize: loss = sum(log(denom) - pos/T) / 8192.
// ---------------------------------------------------------------------------

typedef __attribute__((ext_vector_type(16))) _Float16 v16h;
typedef __attribute__((ext_vector_type(8)))  _Float16 v8h;
typedef __attribute__((ext_vector_type(4)))  _Float16 v4h;
typedef __attribute__((ext_vector_type(8)))  float    v8f;

#define BS        4096      // B*N
#define TWO_BS    8192
#define DIM       512
#define INV_T     2.0f      // 1 / 0.5

#define LDS_ROW   520       // 512 halves + 8 pad -> 1040B stride, conflict-free
#define BUF_HALF  (64 * LDS_ROW)
#define BUF_BYTES (BUF_HALF * 2)          // 66560 B
#define SMEM_BYTES (2 * BUF_BYTES)        // ~130 KB of 320 KB WGP LDS

// ------------------- Kernel 1: row L2-normalize -> f16 ---------------------
__global__ void nt_normalize(const float* __restrict__ inst,
                             const float* __restrict__ bag,
                             _Float16* __restrict__ Zh) {
  const int wave = threadIdx.x >> 5;
  const int lane = threadIdx.x & 31;
  const int row  = blockIdx.x * 8 + wave;

  const float* src = (row < BS) ? (inst + (size_t)row * DIM)
                                : (bag  + (size_t)((row - BS) >> 9) * DIM);
  const float4* s4 = (const float4*)src;

  float4 x[4];
  float ss = 0.f;
#pragma unroll
  for (int i = 0; i < 4; ++i) {
    x[i] = s4[lane + i * 32];
    ss += x[i].x * x[i].x + x[i].y * x[i].y + x[i].z * x[i].z + x[i].w * x[i].w;
  }
#pragma unroll
  for (int off = 16; off >= 1; off >>= 1) ss += __shfl_xor(ss, off, 32);

  const float scale = 1.0f / fmaxf(sqrtf(ss), 1e-12f);

  _Float16* drow = Zh + (size_t)row * DIM;
#pragma unroll
  for (int i = 0; i < 4; ++i) {
    v4h h;
    h[0] = (_Float16)(x[i].x * scale);
    h[1] = (_Float16)(x[i].y * scale);
    h[2] = (_Float16)(x[i].z * scale);
    h[3] = (_Float16)(x[i].w * scale);
    *(v4h*)(drow + (size_t)(lane + i * 32) * 4) = h;
  }
}

// --------- Kernel 2: fused Z Z^T GEMM + masked exp row-sum epilogue --------
// Block = 256 threads = 8 waves; wave w owns rows [blk.x*128 + w*16, +16).
// Grid = (64 row tiles, 8 column splits); block covers 1024 columns in
// 16 tiles of 64 cols; each 64KB B-slab async-staged to LDS, double buffered.
__global__ void __launch_bounds__(256, 1)
nt_sim(const _Float16* __restrict__ Zh,
       float* __restrict__ rowsum,
       float* __restrict__ pos) {
  extern __shared__ _Float16 smem[];   // [2][64][LDS_ROW]

  const int wave    = threadIdx.x >> 5;
  const int lane    = threadIdx.x & 31;
  const int laneM   = lane & 15;
  const int halfSel = lane >> 4;
  const int rowBase = blockIdx.x * 128 + wave * 16;

  const unsigned smemBase = (unsigned)(uintptr_t)&smem[0];

  // Stage one B tile (64 contiguous Z rows = 64KB slab) into LDS bufSel.
  // 16 async b128 per wave (each moves 32 lanes x 16B) -> ASYNCcnt += 16.
  auto issueCopy = [&](int bufSel, int colBase) {
    const _Float16* gbase = Zh + (size_t)colBase * DIM;
    const unsigned ldsBuf = smemBase + (unsigned)bufSel * BUF_BYTES;
    const int t = threadIdx.x;
#pragma unroll
    for (int i = 0; i < 16; ++i) {
      const int idx = t + i * 256;     // 0..4095 16B-chunks
      const int r   = idx >> 6;        // tile row (64 chunks per row)
      const int c   = idx & 63;        // chunk within row
      const _Float16* g = gbase + r * DIM + c * 8;
      const unsigned  l = ldsBuf + (unsigned)(r * (LDS_ROW * 2) + c * 16);
      asm volatile("global_load_async_to_lds_b128 %0, %1, off"
                   :: "v"(l), "v"(g) : "memory");
    }
  };

  // Preload A strip (16 rows x K=512) as 16 register-resident fragments.
  v16h afrag[16];
  const _Float16* arow = Zh + (size_t)(rowBase + laneM) * DIM;
#pragma unroll
  for (int k = 0; k < 16; ++k) {
    union { v16h v; v8h h[2]; } au;
    au.h[0] = *(const v8h*)(arow + k * 32 + halfSel * 8);
    au.h[1] = *(const v8h*)(arow + k * 32 + 16 + halfSel * 8);
    afrag[k] = au.v;
  }

  float partial[8];
#pragma unroll
  for (int v = 0; v < 8; ++v) partial[v] = 0.f;

  const int ct0 = blockIdx.y * 16;
  issueCopy(0, ct0 * 64);              // prime the pipeline

  for (int ci = 0; ci < 16; ++ci) {
    const int cur     = ci & 1;
    const int colBase = (ct0 + ci) * 64;

    if (ci + 1 < 16) {
      issueCopy(cur ^ 1, (ct0 + ci + 1) * 64);        // fill next buffer
      asm volatile("s_wait_asynccnt 16" ::: "memory"); // cur's 16 complete
    } else {
      asm volatile("s_wait_asynccnt 0" ::: "memory");
    }
    __syncthreads();                    // all waves' copies for cur visible

    const _Float16* bt = smem + (size_t)cur * BUF_HALF;

    // j-outer / k-inner: each cacc[j] chain is serial through its C operand,
    // so the ds_load for frag k+1 issues under WMMA k and its s_wait_dscnt
    // is already satisfied — LDS latency hidden behind matrix-pipe latency.
#pragma unroll
    for (int j = 0; j < 4; ++j) {
      const _Float16* brow = bt + (size_t)(j * 16 + laneM) * LDS_ROW
                                + halfSel * 8;
      union { v16h v; v8h h[2]; } bbuf[2];
      bbuf[0].h[0] = *(const v8h*)(brow);
      bbuf[0].h[1] = *(const v8h*)(brow + 16);

      v8f c = (v8f){};
#pragma unroll
      for (int k = 0; k < 16; ++k) {
        if (k + 1 < 16) {
          const _Float16* bp = brow + (k + 1) * 32;
          bbuf[(k + 1) & 1].h[0] = *(const v8h*)(bp);
          bbuf[(k + 1) & 1].h[1] = *(const v8h*)(bp + 16);
        }
        c = __builtin_amdgcn_wmma_f32_16x16x32_f16(
            false, afrag[k], false, bbuf[k & 1].v, (short)0, c, false, false);
      }

      // Epilogue for this 16x16 tile.
      // C layout: lane holds N=lane%16; VGPR v holds M = v + 8*(lane/16).
      const int col = colBase + j * 16 + laneM;
#pragma unroll
      for (int v = 0; v < 8; ++v) {
        const int row = rowBase + v + halfSel * 8;
        const float s = c[v];
        float e = __expf(s * INV_T);
        if (col == row) e = 0.f;                       // self-similarity mask
        partial[v] += e;
        const int mate = (row < BS) ? row + BS : row - BS;
        if (col == mate) pos[row] = s;                 // positive pair (once)
      }
    }

    __syncthreads();   // all waves done reading cur before it is refilled
  }

  // Reduce partial sums across the 16 lanes of each half; one atomic per row.
#pragma unroll
  for (int v = 0; v < 8; ++v) {
    float s = partial[v];
#pragma unroll
    for (int off = 8; off >= 1; off >>= 1) s += __shfl_xor(s, off, 32);
    if (laneM == 0) atomicAdd(&rowsum[rowBase + v + halfSel * 8], s);
  }
}

// ------------------- Kernel 3: scalar loss reduction -----------------------
__global__ void nt_finalize(const float* __restrict__ rowsum,
                            const float* __restrict__ pos,
                            float* __restrict__ out) {
  float local = 0.f;
  for (int i = threadIdx.x; i < TWO_BS; i += 256)
    local += __logf(rowsum[i]) - pos[i] * INV_T;   // -(pos/T - log(denom))

#pragma unroll
  for (int off = 16; off >= 1; off >>= 1) local += __shfl_xor(local, off, 32);

  __shared__ float red[8];
  if ((threadIdx.x & 31) == 0) red[threadIdx.x >> 5] = local;
  __syncthreads();
  if (threadIdx.x < 32) {
    float s = (threadIdx.x < 8) ? red[threadIdx.x] : 0.f;
#pragma unroll
    for (int off = 4; off >= 1; off >>= 1) s += __shfl_xor(s, off, 32);
    if (threadIdx.x == 0) out[0] = s / (float)TWO_BS;
  }
}

// ---------------------------------------------------------------------------
extern "C" void kernel_launch(void* const* d_in, const int* in_sizes, int n_in,
                              void* d_out, int out_size, void* d_ws, size_t ws_size,
                              hipStream_t stream) {
  (void)in_sizes; (void)n_in; (void)out_size; (void)ws_size;

  const float* inst = (const float*)d_in[0];   // [8,512,512] f32
  const float* bag  = (const float*)d_in[1];   // [8,512]     f32
  float* out = (float*)d_out;                  // scalar f32

  // Workspace: f16 Z [8192x512] (8 MB) | rowsum f32[8192] | pos f32[8192]
  _Float16* Zh = (_Float16*)d_ws;
  const size_t zbytes = (size_t)TWO_BS * DIM * sizeof(_Float16);
  float* rowsum = (float*)((char*)d_ws + zbytes);
  float* pos    = rowsum + TWO_BS;

  hipMemsetAsync(rowsum, 0, TWO_BS * sizeof(float), stream);

  nt_normalize<<<dim3(TWO_BS / 8), dim3(256), 0, stream>>>(inst, bag, Zh);
  nt_sim<<<dim3(TWO_BS / 128, 8), dim3(256), SMEM_BYTES, stream>>>(Zh, rowsum, pos);
  nt_finalize<<<dim3(1), dim3(256), 0, stream>>>(rowsum, pos, out);
}
